// RBFLayer_53300544143585
// MI455X (gfx1250) — compile-verified, matrix operand order checked by reading, not run on hardware
//
#include <hip/hip_runtime.h>

// Problem sizes (fixed by the reference).
#define B_DIM 8192
#define I_DIM 512
#define C_DIM 2048
#define KT    (I_DIM / 32)   // 16  K-tiles of 32
#define CT    (C_DIM / 16)   // 128 column tiles of 16
#define MT    (B_DIM / 16)   // 512 row tiles of 16

// GEMM block tile: 64 rows (4 mt) x 128 cols (8 ct), 8 waves of 32.
#define BLK_MT 4
#define BLK_CT 8
#define STAGE_BYTES  (BLK_MT * 4 * 1024 + BLK_CT * 4 * 1024)  // 16KB A + 32KB B = 48KB
#define A_REGION     0
#define B_REGION     (BLK_MT * 4 * 1024)                      // 16KB

typedef __attribute__((ext_vector_type(16))) __bf16 v16bf;
typedef __attribute__((ext_vector_type(8)))  float  v8f;
typedef __attribute__((ext_vector_type(4)))  unsigned int u32x4;
typedef __attribute__((ext_vector_type(8)))  int i32x8;
typedef __attribute__((ext_vector_type(4)))  int i32x4;

// -------- helpers ------------------------------------------------------------

__device__ __forceinline__ void split2(float v, __bf16& hi, __bf16& lo) {
    hi = (__bf16)v;                 // RNE to bf16 (top ~8 mantissa bits)
    lo = (__bf16)(v - (float)hi);   // residual (next ~8 bits)
}

// Element offset inside a 32x16 (or 16x32) 16-bit WMMA fragment, per CDNA5 ISA:
// lanes 0-15 hold K = {0..7, 16..23}, lanes 16-31 hold K = {8..15, 24..31}.
__device__ __forceinline__ int frag_off(int rc, int k) {
    int hi_group = ((k & 15) >= 8) ? 1 : 0;
    int lane = rc + 16 * hi_group;
    int kk = k - 8 * hi_group;
    int e = (kk < 8) ? kk : (kk - 8);
    return lane * 16 + e;
}

__device__ __forceinline__ v8f wmma_bf16(v16bf a, v16bf b, v8f c) {
    return __builtin_amdgcn_wmma_f32_16x16x32_bf16(false, a, false, b,
                                                   (short)0, c, false, false);
}

// TDM 2D tile load: nrows rows of 1024 bytes, global row stride KT*1024 bytes,
// packed contiguously into LDS at lds_addr. data_size = 4 bytes.
__device__ __forceinline__ void tdm_load_tile(const void* gptr,
                                              unsigned int lds_addr,
                                              unsigned int nrows) {
    unsigned long long ga = (unsigned long long)gptr;

    u32x4 g0;
    g0[0] = 1u;                                   // count=1 valid, user desc
    g0[1] = lds_addr;                             // D#.lds_addr (bytes)
    g0[2] = (unsigned int)ga;                     // global_addr[31:0]
    g0[3] = (unsigned int)((ga >> 32) & 0x01FFFFFFu) | 0x80000000u; // [56:32]|type=2

    const unsigned int TD0   = 256u;              // tensor_dim0 (dwords per row)
    const unsigned int TILE0 = 256u;              // tile_dim0
    const unsigned int STR0  = (KT * 1024u) / 4u; // tensor_dim0_stride = 4096 dw

    i32x8 g1;
    g1[0] = (int)(2u << 16);                      // data_size = 4B; mask/flags 0
    g1[1] = (int)((TD0 & 0xFFFFu) << 16);         // tensor_dim0[15:0]
    g1[2] = (int)((TD0 >> 16) | ((nrows & 0xFFFFu) << 16)); // td0 hi | td1 lo
    g1[3] = (int)((nrows >> 16) | (TILE0 << 16)); // td1 hi | tile_dim0
    g1[4] = (int)(nrows & 0xFFFFu);               // tile_dim1 | tile_dim2=0
    g1[5] = (int)STR0;                            // tensor_dim0_stride[31:0]
    g1[6] = 0;                                    // stride0 hi | stride1 lo
    g1[7] = 0;                                    // stride1 hi

    i32x4 z4 = {0, 0, 0, 0};                      // groups 2/3: unused (2D tile)
    i32x8 z8 = {0, 0, 0, 0, 0, 0, 0, 0};          // extra arg (clang-23 form)
    // 6-arg toolchain signature: (g0, g1, g2, g3, <i32x8>, cpol)
    __builtin_amdgcn_tensor_load_to_lds(g0, g1, z4, z4, z8, 0);
}

// -------- prep: A-side (x^2 and x, hi/lo bf16, fragment-swizzled) ------------

__global__ __launch_bounds__(256) void rbf_build_a(
    const float* __restrict__ x,
    __bf16* __restrict__ x2h, __bf16* __restrict__ x2l,
    __bf16* __restrict__ xh,  __bf16* __restrict__ xl) {
    int idx = blockIdx.x * 256 + threadIdx.x;          // over B*I
    if (idx >= B_DIM * I_DIM) return;
    int b = idx / I_DIM;
    int i = idx - b * I_DIM;
    float v = x[idx];

    int mt = b >> 4, m = b & 15, kt = i >> 5, k = i & 31;
    size_t off = (size_t)(mt * KT + kt) * 512 + frag_off(m, k);

    __bf16 h, l;
    split2(v * v, h, l); x2h[off] = h; x2l[off] = l;
    split2(v,     h, l); xh[off]  = h; xl[off]  = l;
}

// -------- prep: B-side (P = w^2, Q = -2 w^2 c, hi/lo, fragment-swizzled) -----

__global__ __launch_bounds__(256) void rbf_build_b(
    const float* __restrict__ weight,   // (I, C) row-major
    const float* __restrict__ centers,  // (C, I) row-major
    __bf16* __restrict__ ph, __bf16* __restrict__ pl,
    __bf16* __restrict__ qh, __bf16* __restrict__ ql) {
    int idx = blockIdx.x * 256 + threadIdx.x;          // over I*C
    if (idx >= I_DIM * C_DIM) return;
    int i = idx / C_DIM;
    int c = idx - i * C_DIM;

    float w  = weight[idx];
    float cv = centers[(size_t)c * I_DIM + i];
    float w2 = w * w;

    int ct = c >> 4, n = c & 15, kt = i >> 5, k = i & 31;
    size_t off = (size_t)(ct * KT + kt) * 512 + frag_off(n, k);

    __bf16 h, l;
    split2(w2,              h, l); ph[off] = h; pl[off] = l;
    split2(-2.0f * w2 * cv, h, l); qh[off] = h; ql[off] = l;
}

// -------- prep: t3[c] = sum_i w2[i,c] * centers[c,i]^2 (f32, exact) ----------

__global__ __launch_bounds__(256) void rbf_t3(
    const float* __restrict__ weight,
    const float* __restrict__ centers,
    float* __restrict__ t3) {
    int wv   = threadIdx.x >> 5;
    int lane = threadIdx.x & 31;
    int c = blockIdx.x * 8 + wv;
    if (c >= C_DIM) return;

    float s = 0.0f;
    for (int i = lane; i < I_DIM; i += 32) {
        float w  = weight[(size_t)i * C_DIM + c];
        float cv = centers[(size_t)c * I_DIM + i];
        s += (w * w) * (cv * cv);
    }
    for (int off = 16; off > 0; off >>= 1)
        s += __shfl_xor(s, off, 32);
    if (lane == 0) t3[c] = s;
}

// -------- main fused GEMM: out = X2@P + X@Q + t3  --------------------------
// TDM double-buffered LDS staging; bf16x3 split-precision WMMA.

__global__ __launch_bounds__(256) void rbf_gemm(
    const __bf16* __restrict__ x2h, const __bf16* __restrict__ x2l,
    const __bf16* __restrict__ xh,  const __bf16* __restrict__ xl,
    const __bf16* __restrict__ ph,  const __bf16* __restrict__ pl,
    const __bf16* __restrict__ qh,  const __bf16* __restrict__ ql,
    const float* __restrict__ t3, float* __restrict__ out) {

    __shared__ char smem[2 * STAGE_BYTES];        // 96 KB double buffer

    const int lane = threadIdx.x & 31;
    const int wv   = threadIdx.x >> 5;            // 8 waves
    const int wm   = wv >> 2;                     // 0..1 : row half
    const int wn   = wv & 3;                      // 0..3 : col quarter
    const int mt0  = blockIdx.x * BLK_MT;         // 4 row tiles / block
    const int ct0  = blockIdx.y * BLK_CT;         // 8 col tiles / block

    // Per-wave TDM descriptor: wave wv stages array wv.
    //   wv 0..3 -> A arrays {x2h,x2l,xh,xl}: 4 rows  -> A_REGION + wv*4KB
    //   wv 4..7 -> B arrays {ph,pl,qh,ql}:   8 rows  -> B_REGION + (wv-4)*8KB
    const __bf16* garr =
        (wv == 0) ? x2h : (wv == 1) ? x2l : (wv == 2) ? xh : (wv == 3) ? xl :
        (wv == 4) ? ph  : (wv == 5) ? pl  : (wv == 6) ? qh : ql;
    const unsigned int nrows   = (wv < 4) ? BLK_MT : BLK_CT;
    const int          tilebase= (wv < 4) ? mt0 : ct0;
    const unsigned int ldsoff  = (wv < 4) ? (A_REGION + wv * 4096u)
                                          : (B_REGION + (wv - 4) * 8192u);

    v8f acc[2][2] = {};

    // Stage kt=0 into buffer 0.
    {
        const char* g = (const char*)garr + (size_t)(tilebase * KT + 0) * 1024;
        unsigned int lds = (unsigned int)(unsigned long long)(smem + ldsoff);
        tdm_load_tile(g, lds, nrows);
        __builtin_amdgcn_s_wait_tensorcnt(0);
    }
    __syncthreads();

    for (int kt = 0; kt < KT; ++kt) {
        const int phase = kt & 1;

        // Kick off next stage (overlaps with compute below).
        if (kt + 1 < KT) {
            const char* g = (const char*)garr +
                            (size_t)(tilebase * KT + (kt + 1)) * 1024;
            unsigned int lds = (unsigned int)(unsigned long long)
                               (smem + (phase ^ 1) * STAGE_BYTES + ldsoff);
            tdm_load_tile(g, lds, nrows);
        }

        // ---- compute from current stage ----
        const char* s = smem + phase * STAGE_BYTES;

        // 8 B fragments for this wave's 2 column tiles.
        v16bf fb[2][4];
#pragma unroll
        for (int jn = 0; jn < 2; ++jn) {
            int ctl = wn * 2 + jn;
#pragma unroll
            for (int b = 0; b < 4; ++b)
                fb[jn][b] = *((const v16bf*)(s + B_REGION + b * 8192 +
                                             ctl * 1024) + lane);
        }

#pragma unroll
        for (int im = 0; im < 2; ++im) {
            int mtl = wm * 2 + im;
            v16bf fa_x2h = *((const v16bf*)(s + A_REGION + 0 * 4096 + mtl * 1024) + lane);
            v16bf fa_x2l = *((const v16bf*)(s + A_REGION + 1 * 4096 + mtl * 1024) + lane);
            v16bf fa_xh  = *((const v16bf*)(s + A_REGION + 2 * 4096 + mtl * 1024) + lane);
            v16bf fa_xl  = *((const v16bf*)(s + A_REGION + 3 * 4096 + mtl * 1024) + lane);
#pragma unroll
            for (int jn = 0; jn < 2; ++jn) {
                // (X2h+X2l)(Ph+Pl): hi*hi + hi*lo + lo*hi
                acc[im][jn] = wmma_bf16(fa_x2h, fb[jn][0], acc[im][jn]);
                acc[im][jn] = wmma_bf16(fa_x2h, fb[jn][1], acc[im][jn]);
                acc[im][jn] = wmma_bf16(fa_x2l, fb[jn][0], acc[im][jn]);
                // (Xh+Xl)(Qh+Ql): hi*hi + hi*lo + lo*hi
                acc[im][jn] = wmma_bf16(fa_xh,  fb[jn][2], acc[im][jn]);
                acc[im][jn] = wmma_bf16(fa_xh,  fb[jn][3], acc[im][jn]);
                acc[im][jn] = wmma_bf16(fa_xl,  fb[jn][2], acc[im][jn]);
            }
        }

        // Make next stage visible to all waves before anyone reads it.
        if (kt + 1 < KT)
            __builtin_amdgcn_s_wait_tensorcnt(0);
        __syncthreads();
    }

    // Epilogue: C/D layout -> lane holds column n = lane%16, rows r + 8*(lane/16).
    const int n    = lane & 15;
    const int half = lane >> 4;
#pragma unroll
    for (int im = 0; im < 2; ++im) {
#pragma unroll
        for (int jn = 0; jn < 2; ++jn) {
            int c = (ct0 + wn * 2 + jn) * 16 + n;
            float t = t3[c];
#pragma unroll
            for (int r = 0; r < 8; ++r) {
                int row = (mt0 + wm * 2 + im) * 16 + half * 8 + r;
                out[(size_t)row * C_DIM + c] = acc[im][jn][r] + t;
            }
        }
    }
}

// -------- launch -------------------------------------------------------------

extern "C" void kernel_launch(void* const* d_in, const int* in_sizes, int n_in,
                              void* d_out, int out_size, void* d_ws, size_t ws_size,
                              hipStream_t stream) {
    const float* x       = (const float*)d_in[0];   // (B, I)
    const float* weight  = (const float*)d_in[1];   // (I, C)
    const float* centers = (const float*)d_in[2];   // (C, I)
    float* out = (float*)d_out;

    char* ws = (char*)d_ws;
    const size_t szA = (size_t)B_DIM * I_DIM * sizeof(__bf16);  // 8 MB each
    const size_t szB = (size_t)I_DIM * C_DIM * sizeof(__bf16);  // 2 MB each

    __bf16* x2h = (__bf16*)(ws);
    __bf16* x2l = (__bf16*)(ws + szA);
    __bf16* xh  = (__bf16*)(ws + 2 * szA);
    __bf16* xl  = (__bf16*)(ws + 3 * szA);
    __bf16* ph  = (__bf16*)(ws + 4 * szA);
    __bf16* pl  = (__bf16*)(ws + 4 * szA + szB);
    __bf16* qh  = (__bf16*)(ws + 4 * szA + 2 * szB);
    __bf16* ql  = (__bf16*)(ws + 4 * szA + 3 * szB);
    float*  t3  = (float*)(ws + 4 * szA + 4 * szB);             // + 8 KB

    rbf_build_a<<<(B_DIM * I_DIM) / 256, 256, 0, stream>>>(x, x2h, x2l, xh, xl);
    rbf_build_b<<<(I_DIM * C_DIM) / 256, 256, 0, stream>>>(weight, centers,
                                                           ph, pl, qh, ql);
    rbf_t3<<<C_DIM / 8, 256, 0, stream>>>(weight, centers, t3);

    dim3 grid(MT / BLK_MT, CT / BLK_CT);   // 128 x 16 blocks
    rbf_gemm<<<grid, 256, 0, stream>>>(x2h, x2l, xh, xl,
                                       ph, pl, qh, ql, t3, out);
}
